// GraphNetwork_11338713661556
// MI455X (gfx1250) — matrix operand order repaired and checked by reading, not compiled, not run on hardware
//
#include <hip/hip_runtime.h>
#include <hip/hip_bf16.h>

typedef __bf16 bf16_t;
typedef __attribute__((ext_vector_type(16))) __bf16 v16bf;
typedef __attribute__((ext_vector_type(4)))  __bf16 bf16x4;
typedef __attribute__((ext_vector_type(8)))  float  v8f;
typedef __attribute__((ext_vector_type(4)))  float  f32x4;

#define DIM 64
#define K_EDGE 192   // 3*D
#define K_NODE 128   // 2*D
#define EDGE_TILES_PER_WG 8

// ---------------------------------------------------------------- utilities
__global__ void zero_kernel(float* __restrict__ p, long long n) {
    long long i = (long long)blockIdx.x * blockDim.x + threadIdx.x;
    long long stride = (long long)gridDim.x * blockDim.x;
    for (; i < n; i += stride) p[i] = 0.0f;
}

__global__ void count_kernel(const int* __restrict__ receivers,
                             float* __restrict__ counts, int n_edges) {
    int i = blockIdx.x * blockDim.x + threadIdx.x;
    if (i < n_edges) atomicAdd(&counts[receivers[i]], 1.0f);
}

// Pack [L, K, 64] f32 weights into per-lane WMMA B-fragment order (bf16) so a
// lane can load its whole v16bf fragment as one contiguous 32B read.
// dst = (((l*KC + kb)*2 + laneHi)*64 + ncol)*16 + t  <=  k = kb*32+laneHi*16+t
__global__ void pack_weights_kernel(const float* __restrict__ W,
                                    bf16_t* __restrict__ Wp,
                                    int KC, int total) {
    int dst = blockIdx.x * blockDim.x + threadIdx.x;
    if (dst >= total) return;
    int t      = dst & 15;
    int ncol   = (dst >> 4) & 63;
    int rest   = dst >> 10;
    int laneHi = rest & 1;
    rest >>= 1;
    int kb = rest % KC;
    int l  = rest / KC;
    int K  = KC * 32;
    int k  = kb * 32 + laneHi * 16 + t;
    Wp[dst] = (bf16_t)W[((size_t)l * K + k) * DIM + ncol];
}

// ---------------------------------------------------------------- EdgeBlock
// out_edges[e,:] = relu(concat(edges[e], nodes[snd[e]], nodes[rcv[e]]) @ W + b)
// fused: atomicAdd(agg[rcv[e], :], out_edges[e,:])   (L2-resident atomics)
__global__ __launch_bounds__(128)
void edge_block_kernel(const float* __restrict__ edges_in,
                       const float* __restrict__ nodes,
                       const int*   __restrict__ senders,
                       const int*   __restrict__ receivers,
                       const bf16_t* __restrict__ Wpk,    // packed fragments
                       const float* __restrict__ bias,    // [64]
                       float* __restrict__ edges_out,
                       float* __restrict__ agg,
                       int n_edges)
{
    __shared__ __align__(16) bf16_t ldsA[16 * K_EDGE];
    __shared__ int    ldsS[16];
    __shared__ int    ldsR[16];
    __shared__ __align__(16) float  ldsOut[16 * DIM];

    const int tid  = threadIdx.x;
    const int wave = tid >> 5;
    const int lane = tid & 31;
    const int ncol = wave * 16 + (lane & 15);
    const int laneHi = lane >> 4;

    // Each lane loads its 6 pre-packed B fragments as contiguous 32B reads.
    v16bf bfrag[6];
#pragma unroll
    for (int kb = 0; kb < 6; ++kb) {
        bfrag[kb] = *(const v16bf*)(Wpk + ((size_t)(kb * 2 + laneHi) * 64 + ncol) * 16);
    }
    const float bcol = bias[ncol];

    for (int t = 0; t < EDGE_TILES_PER_WG; ++t) {
        const int e0 = (blockIdx.x * EDGE_TILES_PER_WG + t) * 16;
        if (e0 >= n_edges) break;

        if (tid < 16) {
            int e = e0 + tid;
            ldsS[tid] = (e < n_edges) ? senders[e] : 0;
        } else if (tid < 32) {
            int e = e0 + (tid - 16);
            ldsR[tid - 16] = (e < n_edges) ? receivers[e] : 0;
        }
        __syncthreads();

        // Vectorized gather: float4 global loads -> bf16x4 LDS stores.
        for (int idx = tid; idx < 16 * (K_EDGE / 4); idx += 128) {
            int m = idx / (K_EDGE / 4);
            int q = idx - m * (K_EDGE / 4);     // float4 index within row
            int e = e0 + m;
            f32x4 val = {};
            if (e < n_edges) {
                if (q < 16)      val = *(const f32x4*)(edges_in + (size_t)e       * DIM + q * 4);
                else if (q < 32) val = *(const f32x4*)(nodes + (size_t)ldsS[m] * DIM + (q - 16) * 4);
                else             val = *(const f32x4*)(nodes + (size_t)ldsR[m] * DIM + (q - 32) * 4);
            }
            bf16x4 bv;
            bv[0] = (bf16_t)val[0]; bv[1] = (bf16_t)val[1];
            bv[2] = (bf16_t)val[2]; bv[3] = (bf16_t)val[3];
            *(bf16x4*)&ldsA[m * K_EDGE + q * 4] = bv;
        }
        __syncthreads();

        // 6 x v_wmma_f32_16x16x32_bf16 per 16x16 output tile.
        v8f c = {};
        const int mrow = lane & 15;
#pragma unroll
        for (int kb = 0; kb < 6; ++kb) {
            v16bf afrag;
#pragma unroll
            for (int v = 0; v < 8; ++v) {
                // A k order used here, matching packed B: VGPR v -> K = 16*laneHi + 2*v
                int k0 = kb * 32 + laneHi * 16 + v * 2;
                afrag[2 * v]     = ldsA[mrow * K_EDGE + k0];
                afrag[2 * v + 1] = ldsA[mrow * K_EDGE + k0 + 1];
            }
            c = __builtin_amdgcn_wmma_f32_16x16x32_bf16(
                    false, afrag, false, bfrag[kb], (short)0, c, false, false);
        }

        // bias + relu, stage in LDS (D layout: VGPR v -> row 8*laneHi + v)
#pragma unroll
        for (int v = 0; v < 8; ++v) {
            int m = laneHi * 8 + v;
            float val = c[v] + bcol;
            ldsOut[m * DIM + ncol] = val > 0.0f ? val : 0.0f;
        }
        __syncthreads();

        // Vectorized coalesced store + scatter-add into L2-resident agg.
        for (int idx = tid; idx < 16 * (DIM / 4); idx += 128) {
            int m = idx >> 4, q = idx & 15;
            int e = e0 + m;
            if (e < n_edges) {
                f32x4 v = *(const f32x4*)&ldsOut[m * DIM + q * 4];
                *(f32x4*)(edges_out + (size_t)e * DIM + q * 4) = v;
                float* ap = agg + (size_t)ldsR[m] * DIM + q * 4;
                atomicAdd(ap + 0, v[0]);
                atomicAdd(ap + 1, v[1]);
                atomicAdd(ap + 2, v[2]);
                atomicAdd(ap + 3, v[3]);
            }
        }
        __syncthreads();
    }
}

// ---------------------------------------------------------------- NodeBlock
// nodes_out = relu(concat(nodes, agg/max(count,1)) @ W + b)
__global__ __launch_bounds__(128)
void node_block_kernel(const float* __restrict__ nodes_in,
                       const float* __restrict__ agg,
                       const float* __restrict__ counts,
                       const bf16_t* __restrict__ Wpk,   // packed fragments
                       const float* __restrict__ bias,   // [64]
                       float* __restrict__ nodes_out,
                       int n_nodes)
{
    __shared__ __align__(16) bf16_t ldsA[16 * K_NODE];
    __shared__ __align__(16) float  ldsOut[16 * DIM];

    const int tid  = threadIdx.x;
    const int wave = tid >> 5;
    const int lane = tid & 31;
    const int ncol = wave * 16 + (lane & 15);
    const int laneHi = lane >> 4;
    const int n0 = blockIdx.x * 16;

    v16bf bfrag[4];
#pragma unroll
    for (int kb = 0; kb < 4; ++kb) {
        bfrag[kb] = *(const v16bf*)(Wpk + ((size_t)(kb * 2 + laneHi) * 64 + ncol) * 16);
    }
    const float bcol = bias[ncol];

    // Vectorized gather: [nodes | agg * (1/clamped count)]
    for (int idx = tid; idx < 16 * (K_NODE / 4); idx += 128) {
        int m = idx >> 5;
        int q = idx & 31;
        int node = n0 + m;
        f32x4 val = {};
        if (node < n_nodes) {
            if (q < 16) {
                val = *(const f32x4*)(nodes_in + (size_t)node * DIM + q * 4);
            } else {
                float cnt = counts[node];
                cnt = cnt < 1.0f ? 1.0f : cnt;
                float inv = 1.0f / cnt;
                f32x4 a = *(const f32x4*)(agg + (size_t)node * DIM + (q - 16) * 4);
                val[0] = a[0] * inv; val[1] = a[1] * inv;
                val[2] = a[2] * inv; val[3] = a[3] * inv;
            }
        }
        bf16x4 bv;
        bv[0] = (bf16_t)val[0]; bv[1] = (bf16_t)val[1];
        bv[2] = (bf16_t)val[2]; bv[3] = (bf16_t)val[3];
        *(bf16x4*)&ldsA[m * K_NODE + q * 4] = bv;
    }
    __syncthreads();

    v8f c = {};
    const int mrow = lane & 15;
#pragma unroll
    for (int kb = 0; kb < 4; ++kb) {
        v16bf afrag;
#pragma unroll
        for (int v = 0; v < 8; ++v) {
            int k0 = kb * 32 + laneHi * 16 + v * 2;
            afrag[2 * v]     = ldsA[mrow * K_NODE + k0];
            afrag[2 * v + 1] = ldsA[mrow * K_NODE + k0 + 1];
        }
        c = __builtin_amdgcn_wmma_f32_16x16x32_bf16(
                false, afrag, false, bfrag[kb], (short)0, c, false, false);
    }

#pragma unroll
    for (int v = 0; v < 8; ++v) {
        int m = laneHi * 8 + v;
        float val = c[v] + bcol;
        ldsOut[m * DIM + ncol] = val > 0.0f ? val : 0.0f;
    }
    __syncthreads();

    for (int idx = tid; idx < 16 * (DIM / 4); idx += 128) {
        int m = idx >> 4, q = idx & 15;
        int node = n0 + m;
        if (node < n_nodes) {
            *(f32x4*)(nodes_out + (size_t)node * DIM + q * 4) =
                *(const f32x4*)&ldsOut[m * DIM + q * 4];
        }
    }
}

// ------------------------------------------------- LayerNorm + MLP + project
__global__ __launch_bounds__(64)
void final_kernel(const float* __restrict__ nodes,
                  const float* __restrict__ gamma, const float* __restrict__ beta,
                  const float* __restrict__ W1, const float* __restrict__ b1,
                  const float* __restrict__ W2, const float* __restrict__ b2,
                  const float* __restrict__ Wp, const float* __restrict__ bp,
                  float* __restrict__ out, int n_nodes)
{
    __shared__ float x[64];
    __shared__ float h[128];
    __shared__ float red[64];

    const int node = blockIdx.x;
    const int t = threadIdx.x;
    if (node >= n_nodes) return;

    float v = nodes[(size_t)node * DIM + t];

    red[t] = v;
    __syncthreads();
    for (int s = 32; s > 0; s >>= 1) { if (t < s) red[t] += red[t + s]; __syncthreads(); }
    float mu = red[0] * (1.0f / 64.0f);
    __syncthreads();

    float d = v - mu;
    red[t] = d * d;
    __syncthreads();
    for (int s = 32; s > 0; s >>= 1) { if (t < s) red[t] += red[t + s]; __syncthreads(); }
    float rstd = rsqrtf(red[0] * (1.0f / 64.0f) + 1e-5f);
    __syncthreads();

    x[t] = d * rstd * gamma[t] + beta[t];
    __syncthreads();

    float h0 = b1[t], h1 = b1[t + 64];
    for (int k = 0; k < 64; ++k) {
        float xv = x[k];
        h0 = fmaf(xv, W1[k * 128 + t], h0);
        h1 = fmaf(xv, W1[k * 128 + t + 64], h1);
    }
    h[t]      = h0 > 0.0f ? h0 : 0.0f;
    h[t + 64] = h1 > 0.0f ? h1 : 0.0f;
    __syncthreads();

    float y = b2[t];
    for (int k = 0; k < 128; ++k) y = fmaf(h[k], W2[k * 64 + t], y);
    red[t] = y * Wp[t];
    __syncthreads();
    for (int s = 32; s > 0; s >>= 1) { if (t < s) red[t] += red[t + s]; __syncthreads(); }
    if (t == 0) out[node] = red[0] + bp[0];
}

// ---------------------------------------------------------------- launcher
extern "C" void kernel_launch(void* const* d_in, const int* in_sizes, int n_in,
                              void* d_out, int out_size, void* d_ws, size_t ws_size,
                              hipStream_t stream) {
    (void)n_in; (void)out_size; (void)ws_size;

    const float* nodes     = (const float*)d_in[0];
    const float* edges     = (const float*)d_in[1];
    const int*   senders   = (const int*)  d_in[2];
    const int*   receivers = (const int*)  d_in[3];
    const float* We        = (const float*)d_in[4];
    const float* be        = (const float*)d_in[5];
    const float* Wn        = (const float*)d_in[6];
    const float* bn        = (const float*)d_in[7];
    const float* gamma     = (const float*)d_in[8];
    const float* beta      = (const float*)d_in[9];
    const float* W1        = (const float*)d_in[10];
    const float* b1        = (const float*)d_in[11];
    const float* W2        = (const float*)d_in[12];
    const float* b2        = (const float*)d_in[13];
    const float* Wp        = (const float*)d_in[14];
    const float* bp        = (const float*)d_in[15];

    const int E = in_sizes[2];
    const int N = in_sizes[0] / DIM;
    const int L = in_sizes[4] / (K_EDGE * DIM);

    char* ws = (char*)d_ws;
    size_t off = 0;
    auto carve = [&](size_t bytes) -> void* {
        void* p = ws + off;
        off = (off + bytes + 255) & ~(size_t)255;
        return p;
    };
    float*  edges_buf = (float*) carve((size_t)E * DIM * sizeof(float));
    float*  agg       = (float*) carve((size_t)N * DIM * sizeof(float));
    float*  nodesA    = (float*) carve((size_t)N * DIM * sizeof(float));
    float*  nodesB    = (float*) carve((size_t)N * DIM * sizeof(float));
    float*  counts    = (float*) carve((size_t)N * sizeof(float));
    bf16_t* Wepk      = (bf16_t*)carve((size_t)L * K_EDGE * DIM * sizeof(bf16_t));
    bf16_t* Wnpk      = (bf16_t*)carve((size_t)L * K_NODE * DIM * sizeof(bf16_t));

    // 1) in-degree counts
    zero_kernel<<<1024, 256, 0, stream>>>(counts, (long long)N);
    count_kernel<<<(E + 255) / 256, 256, 0, stream>>>(receivers, counts, E);

    // 2) weights -> bf16, pre-swizzled into WMMA fragment order
    {
        int ne = L * K_EDGE * DIM;
        pack_weights_kernel<<<(ne + 255) / 256, 256, 0, stream>>>(We, Wepk, K_EDGE / 32, ne);
        int nn = L * K_NODE * DIM;
        pack_weights_kernel<<<(nn + 255) / 256, 256, 0, stream>>>(Wn, Wnpk, K_NODE / 32, nn);
    }

    // 3) message-passing layers
    const float* edges_cur = edges;
    const float* nodes_cur = nodes;
    float* node_bufs[2] = { nodesA, nodesB };

    const int edge_tiles  = (E + 15) / 16;
    const int edge_blocks = (edge_tiles + EDGE_TILES_PER_WG - 1) / EDGE_TILES_PER_WG;
    const int node_blocks = (N + 15) / 16;

    for (int l = 0; l < L; ++l) {
        zero_kernel<<<2048, 256, 0, stream>>>(agg, (long long)N * DIM);

        edge_block_kernel<<<edge_blocks, 128, 0, stream>>>(
            edges_cur, nodes_cur, senders, receivers,
            Wepk + (size_t)l * K_EDGE * DIM, be + l * DIM,
            edges_buf, agg, E);

        float* nodes_next = node_bufs[l & 1];
        node_block_kernel<<<node_blocks, 128, 0, stream>>>(
            nodes_cur, agg, counts,
            Wnpk + (size_t)l * K_NODE * DIM, bn + l * DIM,
            nodes_next, N);

        edges_cur = edges_buf;
        nodes_cur = nodes_next;
    }

    // 4) layernorm + MLP + projection
    final_kernel<<<N, 64, 0, stream>>>(nodes_cur, gamma, beta,
                                       W1, b1, W2, b2, Wp, bp,
                                       (float*)d_out, N);
}
